// MultiHeadAttention_68650757260116
// MI455X (gfx1250) — compile-verified, matrix operand order checked by reading, not run on hardware
//
#include <hip/hip_runtime.h>
#include <cstdint>
#include <cstddef>

// ---------------- types ----------------
typedef __attribute__((ext_vector_type(16))) __bf16 v16bf;
typedef __attribute__((ext_vector_type(8)))  __bf16 v8bf;
typedef __attribute__((ext_vector_type(4)))  __bf16 v4bf;
typedef __attribute__((ext_vector_type(8)))  float  v8f;

union FragBF {
  v16bf v;
  v8bf  h[2];
};

#define S_LEN   2048
#define DMODEL  4096
#define NHEAD   32
#define HDIM    128
#define ATT_SCALE 0.08838834764831845f      // 1/sqrt(128)
#define LN10000_OVER_128 0.07195578415606394f

__device__ __forceinline__ float rmax16(float x) {
  x = fmaxf(x, __shfl_xor(x, 1, 32));
  x = fmaxf(x, __shfl_xor(x, 2, 32));
  x = fmaxf(x, __shfl_xor(x, 4, 32));
  x = fmaxf(x, __shfl_xor(x, 8, 32));
  return x;
}
__device__ __forceinline__ float rsum16(float x) {
  x += __shfl_xor(x, 1, 32);
  x += __shfl_xor(x, 2, 32);
  x += __shfl_xor(x, 4, 32);
  x += __shfl_xor(x, 8, 32);
  return x;
}

__device__ __forceinline__ v8f wmma_bf16(const FragBF& a, const FragBF& b, v8f c) {
  return __builtin_amdgcn_wmma_f32_16x16x32_bf16(
      /*neg_a=*/false, a.v, /*neg_b=*/false, b.v,
      /*c_mod=*/(short)0, c, /*reuse_a=*/false, /*reuse_b=*/false);
}

// ---- gfx1250 async global->LDS copy (16B per lane), tracked on ASYNCcnt ----
__device__ __forceinline__ void async_copy_b128(const void* gaddr, void* lds_ptr) {
  const uint32_t loff = (uint32_t)(uintptr_t)lds_ptr;  // low 32 bits = LDS offset
  asm volatile("global_load_async_to_lds_b128 %0, %1, off"
               :: "v"(loff), "v"(gaddr) : "memory");
}
__device__ __forceinline__ void wait_async0() {
  asm volatile("s_wait_asynccnt 0" ::: "memory");
}

// =====================================================================
// GEMM: Out[m, n] = A[m, :] . W[n, :] + bias[n]   (i.e. A @ W^T + b)
//   M = 2048, N = K = 4096.
//   EPI: 0 = bf16 out, 1 = RoPE (head-index angle) + bf16 out, 2 = f32 out
//   ABF: A operand is bf16 (true, staged via async-to-LDS) or f32 (false)
// Block: 256 thr (8 waves), tile 64x128, BK=32; wave computes 32x32 (2x2 WMMA)
// Register prefetch of next tile overlaps global latency with WMMA.
// =====================================================================
template<int EPI, bool ABF>
__global__ __launch_bounds__(256) void gemm_wmma(const void*  __restrict__ Aptr,
                                                 const float* __restrict__ W,
                                                 const float* __restrict__ bias,
                                                 void*        __restrict__ Out) {
  constexpr int BM = 64, BN = 128, BK = 32;
  constexpr int LA = 48, LB = 48;  // padded LDS strides (96 B: 16B-aligned rows)
  __shared__ __bf16 At[BM * LA];
  __shared__ __bf16 Bt[BN * LB];

  const int t    = threadIdx.x;
  const int lane = t & 31;
  const int w    = t >> 5;
  const int wr   = w >> 2;          // 0..1  (M dir)
  const int wc   = w & 3;           // 0..3  (N dir)
  const int hl   = lane >> 4;       // wave half
  const int ln   = lane & 15;
  const int m0   = blockIdx.x * BM;
  const int n0   = blockIdx.y * BN;

  v8f acc[2][2] = {};

  // ---- register prefetch state ----
  float4 ra[2];        // A tile chunks (f32 path)
  float4 rb[4];        // W tile chunks
  const float* Af = (const float*)Aptr;

  if constexpr (!ABF) {
#pragma unroll
    for (int i = 0; i < 2; ++i) {
      const int cf = t + 256 * i;
      const int row = cf >> 3, kc = (cf & 7) * 4;
      ra[i] = *(const float4*)(Af + (size_t)(m0 + row) * DMODEL + kc);
    }
  }
#pragma unroll
  for (int i = 0; i < 4; ++i) {
    const int cf = t + 256 * i;
    const int row = cf >> 3, kc = (cf & 7) * 4;
    rb[i] = *(const float4*)(W + (size_t)(n0 + row) * DMODEL + kc);
  }

  for (int kb = 0; kb < DMODEL; kb += BK) {
    __syncthreads();                 // prior iteration's fragment reads done
    // ---- commit staged tiles to LDS ----
    if constexpr (ABF) {
      const __bf16* A = (const __bf16*)Aptr;
      const int row = t >> 2, kc = (t & 3) * 8;    // 256 chunks of 8 bf16
      async_copy_b128(A + (size_t)(m0 + row) * DMODEL + kb + kc,
                      &At[row * LA + kc]);
    } else {
#pragma unroll
      for (int i = 0; i < 2; ++i) {
        const int cf = t + 256 * i;
        const int row = cf >> 3, kc = (cf & 7) * 4;
        v4bf b4 = { (__bf16)ra[i].x, (__bf16)ra[i].y, (__bf16)ra[i].z, (__bf16)ra[i].w };
        *(v4bf*)&At[row * LA + kc] = b4;
      }
    }
#pragma unroll
    for (int i = 0; i < 4; ++i) {
      const int cf = t + 256 * i;
      const int row = cf >> 3, kc = (cf & 7) * 4;
      v4bf b4 = { (__bf16)rb[i].x, (__bf16)rb[i].y, (__bf16)rb[i].z, (__bf16)rb[i].w };
      *(v4bf*)&Bt[row * LB + kc] = b4;
    }
    if constexpr (ABF) wait_async0();
    __syncthreads();

    // ---- prefetch next tile into registers (overlaps with WMMA below) ----
    const int kn = kb + BK;
    if (kn < DMODEL) {
      if constexpr (!ABF) {
#pragma unroll
        for (int i = 0; i < 2; ++i) {
          const int cf = t + 256 * i;
          const int row = cf >> 3, kc = (cf & 7) * 4;
          ra[i] = *(const float4*)(Af + (size_t)(m0 + row) * DMODEL + kn + kc);
        }
      }
#pragma unroll
      for (int i = 0; i < 4; ++i) {
        const int cf = t + 256 * i;
        const int row = cf >> 3, kc = (cf & 7) * 4;
        rb[i] = *(const float4*)(W + (size_t)(n0 + row) * DMODEL + kn + kc);
      }
    }

    // ---- fragments + 2x2 WMMA ----
    FragBF a[2], b[2];
#pragma unroll
    for (int i = 0; i < 2; ++i) {
      const __bf16* p = &At[(wr * 32 + i * 16 + ln) * LA + hl * 8];
      a[i].h[0] = *(const v8bf*)p;
      a[i].h[1] = *(const v8bf*)(p + 16);
    }
#pragma unroll
    for (int j = 0; j < 2; ++j) {
      const __bf16* p = &Bt[(wc * 32 + j * 16 + ln) * LB + hl * 8];
      b[j].h[0] = *(const v8bf*)p;
      b[j].h[1] = *(const v8bf*)(p + 16);
    }
#pragma unroll
    for (int i = 0; i < 2; ++i)
#pragma unroll
      for (int j = 0; j < 2; ++j)
        acc[i][j] = wmma_bf16(a[i], b[j], acc[i][j]);
  }

  // ---- epilogue ----
#pragma unroll
  for (int i = 0; i < 2; ++i)
#pragma unroll
    for (int j = 0; j < 2; ++j)
#pragma unroll
      for (int r = 0; r < 8; ++r) {
        const int row = m0 + wr * 32 + i * 16 + r + 8 * hl;
        const int col = n0 + wc * 32 + j * 16 + ln;
        float val = acc[i][j][r] + bias[col];
        if constexpr (EPI == 1) {
          // RoPE: reference angle = head_index * 10000^(-(2i)/128) (no position!)
          const float pv   = __shfl_xor(val, 1, 32);   // even<->odd column pair
          const int   hd   = col & (HDIM - 1);
          const int   hidx = (col >> 7) & (NHEAD - 1);
          const float ang  = (float)hidx * __expf(-(float)(hd & ~1) * LN10000_OVER_128);
          float sn, cs;
          __sincosf(ang, &sn, &cs);
          val = (col & 1) ? (pv * sn + val * cs) : (val * cs - pv * sn);
        }
        if constexpr (EPI == 2) {
          ((float*)Out)[(size_t)row * DMODEL + col] = val;
        } else {
          ((__bf16*)Out)[(size_t)row * DMODEL + col] = (__bf16)val;
        }
      }
}

// =====================================================================
// Flash attention (causal), one block = (head, 128 query rows), 8 waves.
// Double-buffered LDS staging: K tile via async global->LDS (ASYNCcnt),
// V tile via register transpose, both for block kbi+1 issued while
// computing block kbi (16 WMMA / key block per wave).
// =====================================================================
constexpr int LK = 136;  // K tile stride: 272 B rows -> conflict-free b128 reads
constexpr int LV = 48;   // V^T stride
constexpr int LP = 48;   // P scratch stride

__device__ __forceinline__ void flash_issue(const __bf16* __restrict__ K,
                                            const __bf16* __restrict__ V,
                                            int kk, int h, int t,
                                            __bf16* KtBuf, v8bf vd[2]) {
#pragma unroll
  for (int i = 0; i < 2; ++i) {
    const int cf   = t + 256 * i;       // 512 chunks of 8
    const int row  = cf >> 4;           // key 0..31
    const int colc = (cf & 15) * 8;     // dim 0..120
    const size_t g = (size_t)(kk + row) * DMODEL + h * HDIM + colc;
    async_copy_b128(K + g, &KtBuf[row * LK + colc]);   // async K copy
    vd[i] = *(const v8bf*)(V + g);                     // V into registers
  }
}
__device__ __forceinline__ void flash_commitV(__bf16* VTBuf, int t, const v8bf vd[2]) {
#pragma unroll
  for (int i = 0; i < 2; ++i) {
    const int cf   = t + 256 * i;
    const int row  = cf >> 4;
    const int colc = (cf & 15) * 8;
#pragma unroll
    for (int e = 0; e < 8; ++e) {
      const int ee = (e + t) & 7;       // rotate to spread LDS banks
      VTBuf[(colc + ee) * LV + row] = vd[i][ee];
    }
  }
}

__global__ __launch_bounds__(256) void flash_attn(const __bf16* __restrict__ Q,
                                                  const __bf16* __restrict__ K,
                                                  const __bf16* __restrict__ V,
                                                  __bf16*       __restrict__ O) {
  __shared__ __bf16 Kt[2][32 * LK];
  __shared__ __bf16 VT[2][HDIM * LV];
  __shared__ __bf16 Pt[8][16 * LP];

  const int t     = threadIdx.x;
  const int lane  = t & 31;
  const int w     = t >> 5;
  const int hl    = lane >> 4;
  const int ln    = lane & 15;
  const int h     = blockIdx.x >> 4;
  const int q0    = (blockIdx.x & 15) << 7;  // query block start
  const int qrow0 = q0 + w * 16;             // this wave's first query row

  // ---- Q fragments: 16 x 128 bf16, 4 chunks of K=32, kept in registers ----
  FragBF qf[4];
  {
    const __bf16* qbase = Q + (size_t)(qrow0 + ln) * DMODEL + h * HDIM;
#pragma unroll
    for (int c = 0; c < 4; ++c) {
      qf[c].h[0] = *(const v8bf*)(qbase + c * 32 + hl * 8);
      qf[c].h[1] = *(const v8bf*)(qbase + c * 32 + hl * 8 + 16);
    }
  }

  v8f   acc_o[8] = {};
  float m_st[8], l_st[8];
#pragma unroll
  for (int r = 0; r < 8; ++r) { m_st[r] = -__builtin_inff(); l_st[r] = 0.f; }

  const int nkb = (q0 >> 5) + 4;   // key blocks covering keys <= q0+127

  // ---- prologue: stage block 0 into buffer 0 ----
  {
    v8bf vd[2];
    flash_issue(K, V, 0, h, t, Kt[0], vd);
    flash_commitV(VT[0], t, vd);
    wait_async0();
    __syncthreads();
  }

  for (int kbi = 0; kbi < nkb; ++kbi) {
    const int kk  = kbi << 5;
    const int cur = kbi & 1;
    const bool more = (kbi + 1) < nkb;

    // ---- issue next block's staging (overlaps with compute below) ----
    v8bf vd[2];
    if (more) flash_issue(K, V, (kbi + 1) << 5, h, t, Kt[cur ^ 1], vd);

    if (kk <= qrow0 + 15) {               // wave-uniform causal skip
      const __bf16* KtC = Kt[cur];
      const __bf16* VTC = VT[cur];
      // ---- scores S = Q . K^T : two 16x16 tiles over 32 keys ----
      v8f s0 = {}, s1 = {};
#pragma unroll
      for (int c = 0; c < 4; ++c) {
        FragBF b0, b1;
        const __bf16* pk0 = &KtC[ln * LK + c * 32 + hl * 8];
        b0.h[0] = *(const v8bf*)pk0;
        b0.h[1] = *(const v8bf*)(pk0 + 16);
        const __bf16* pk1 = &KtC[(16 + ln) * LK + c * 32 + hl * 8];
        b1.h[0] = *(const v8bf*)pk1;
        b1.h[1] = *(const v8bf*)(pk1 + 16);
        s0 = wmma_bf16(qf[c], b0, s0);
        s1 = wmma_bf16(qf[c], b1, s1);
      }
#pragma unroll
      for (int r = 0; r < 8; ++r) { s0[r] *= ATT_SCALE; s1[r] *= ATT_SCALE; }
      if (kk + 31 > qrow0) {              // diagonal block: mask
#pragma unroll
        for (int r = 0; r < 8; ++r) {
          const int row = qrow0 + r + 8 * hl;
          if (kk + ln > row)      s0[r] = -__builtin_inff();
          if (kk + 16 + ln > row) s1[r] = -__builtin_inff();
        }
      }
      // ---- online softmax (rows live across 16 lanes of each half) ----
      float p0[8], p1[8];
#pragma unroll
      for (int r = 0; r < 8; ++r) {
        const float mx   = rmax16(fmaxf(s0[r], s1[r]));
        const float mn   = fmaxf(m_st[r], mx);
        const float corr = __expf(m_st[r] - mn);
        m_st[r] = mn;
        p0[r] = __expf(s0[r] - mn);
        p1[r] = __expf(s1[r] - mn);
        const float rs = rsum16(p0[r] + p1[r]);
        l_st[r] = l_st[r] * corr + rs;
#pragma unroll
        for (int tt = 0; tt < 8; ++tt) acc_o[tt][r] *= corr;
      }
      // ---- P: C-layout -> A-layout via per-wave LDS scratch ----
#pragma unroll
      for (int r = 0; r < 8; ++r) {
        Pt[w][(r + 8 * hl) * LP + ln]      = (__bf16)p0[r];
        Pt[w][(r + 8 * hl) * LP + 16 + ln] = (__bf16)p1[r];
      }
      asm volatile("s_wait_dscnt 0" ::: "memory");  // same-wave DS ordering
      FragBF pa;
      pa.h[0] = *(const v8bf*)&Pt[w][ln * LP + hl * 8];
      pa.h[1] = *(const v8bf*)&Pt[w][ln * LP + hl * 8 + 16];
      // ---- acc += P . V  (8 output tiles across HD=128) ----
#pragma unroll
      for (int tt = 0; tt < 8; ++tt) {
        FragBF bv;
        const __bf16* pv = &VTC[(tt * 16 + ln) * LV + hl * 8];
        bv.h[0] = *(const v8bf*)pv;
        bv.h[1] = *(const v8bf*)(pv + 16);
        acc_o[tt] = wmma_bf16(pa, bv, acc_o[tt]);
      }
    }

    // ---- commit next block's V transpose, fence async K, barrier ----
    if (more) flash_commitV(VT[cur ^ 1], t, vd);
    wait_async0();
    __syncthreads();
  }

  // ---- normalize + write O[s, h*128+d] as bf16 ----
#pragma unroll
  for (int tt = 0; tt < 8; ++tt)
#pragma unroll
    for (int r = 0; r < 8; ++r) {
      const int row = qrow0 + r + 8 * hl;
      const int col = h * HDIM + tt * 16 + ln;
      O[(size_t)row * DMODEL + col] = (__bf16)(acc_o[tt][r] / l_st[r]);
    }
}

// =====================================================================
// launch
// =====================================================================
extern "C" void kernel_launch(void* const* d_in, const int* in_sizes, int n_in,
                              void* d_out, int out_size, void* d_ws, size_t ws_size,
                              hipStream_t stream) {
  (void)in_sizes; (void)n_in; (void)out_size; (void)ws_size;
  const float* q  = (const float*)d_in[0];
  const float* k  = (const float*)d_in[1];
  const float* v  = (const float*)d_in[2];
  const float* Wq = (const float*)d_in[3];
  const float* bq = (const float*)d_in[4];
  const float* Wk = (const float*)d_in[5];
  const float* bk = (const float*)d_in[6];
  const float* Wv = (const float*)d_in[7];
  const float* bv = (const float*)d_in[8];
  const float* Wo = (const float*)d_in[9];
  const float* bo = (const float*)d_in[10];
  float* out = (float*)d_out;

  // workspace: 4 bf16 planes of [2048 x 4096] = 4 x 16 MB = 64 MB
  char* ws = (char*)d_ws;
  const size_t plane = (size_t)S_LEN * DMODEL * sizeof(__bf16);
  __bf16* Qp = (__bf16*)(ws);
  __bf16* Kp = (__bf16*)(ws + plane);
  __bf16* Vp = (__bf16*)(ws + 2 * plane);
  __bf16* Op = (__bf16*)(ws + 3 * plane);

  dim3 gg(S_LEN / 64, DMODEL / 128);
  gemm_wmma<1, false><<<gg, 256, 0, stream>>>(q, Wq, bq, Qp);   // Q proj + RoPE
  gemm_wmma<1, false><<<gg, 256, 0, stream>>>(k, Wk, bk, Kp);   // K proj + RoPE
  gemm_wmma<0, false><<<gg, 256, 0, stream>>>(v, Wv, bv, Vp);   // V proj
  flash_attn<<<NHEAD * (S_LEN / 128), 256, 0, stream>>>(Qp, Kp, Vp, Op);
  gemm_wmma<2, true><<<gg, 256, 0, stream>>>(Op, Wo, bo, out);  // out proj (f32)
}
